// RNN_32581621907698
// MI455X (gfx1250) — matrix operand order repaired and checked by reading, not compiled, not run on hardware
//
#include <hip/hip_runtime.h>

typedef __attribute__((ext_vector_type(2)))  float        v2f;
typedef __attribute__((ext_vector_type(8)))  float        v8f;
typedef __attribute__((ext_vector_type(16))) __bf16       v16bf;
typedef __attribute__((ext_vector_type(4)))  unsigned int v4u;

union BF16Frag { v16bf v; v4u u[2]; };

#define BB    128      // batch
#define TT    2048     // time steps
#define NIN   64       // input features
#define HH    100      // hidden
#define NOUT  32       // output features
#define NT_H  7        // H N-tiles (112 padded)
#define KP    128      // K padded to 4 bf16-WMMA steps of 32
#define LDSROW 136     // LDS row stride (bf16 elems): 16B-aligned chunks, bank-skewed

#if __has_builtin(__builtin_amdgcn_tanhf)
  #define TANHF(x) __builtin_amdgcn_tanhf(x)
#elif __has_builtin(__builtin_amdgcn_tanh_f32)
  #define TANHF(x) __builtin_amdgcn_tanh_f32(x)
#else
  #define TANHF(x) tanhf(x)
#endif

__device__ __forceinline__ unsigned short f2bf(float f) {
    unsigned u = __builtin_bit_cast(unsigned, f);
    u += 0x7FFFu + ((u >> 16) & 1u);          // round-to-nearest-even
    return (unsigned short)(u >> 16);
}

// ---------------------------------------------------------------------------
// Kernel 0: pack W_hh^T (B[k][n] = W_hh[n][k]) into bf16, zero-padded [112][KP]
// ---------------------------------------------------------------------------
__global__ void pack_whh_bf16(const float* __restrict__ whh,
                              unsigned short* __restrict__ whhb) {
    int idx = blockIdx.x * 256 + threadIdx.x;     // idx = n*KP + k
    if (idx >= NT_H * 16 * KP) return;
    int n = idx / KP, k = idx % KP;
    float f = (n < HH && k < HH) ? whh[n * HH + k] : 0.0f;
    whhb[idx] = f2bf(f);
}

// ---------------------------------------------------------------------------
// Kernel 1: xp[r][h] = sum_i (x+noise)[r][i] * W_ih[h][i], r = b*T+t
// fp32 WMMA 16x16x4. One wave = one 16-row M-tile x ALL 7 N-tiles, so the
// 128 MB of x+noise is read exactly once; W_ih (25 KB) lives in cache.
// ---------------------------------------------------------------------------
__global__ void input_proj(const float* __restrict__ x,
                           const float* __restrict__ noise,
                           const float* __restrict__ wih,
                           float* __restrict__ xp) {
    int lane = threadIdx.x & 31;
    int wave = threadIdx.x >> 5;
    int r0   = (blockIdx.x * 8 + wave) * 16;      // 16-row tile of (b,t)
    int m    = lane & 15, half = lane >> 4;
    size_t arow = (size_t)(r0 + m);

    v8f c[NT_H];
    #pragma unroll
    for (int nt = 0; nt < NT_H; ++nt) c[nt] = (v8f){};

    #pragma unroll
    for (int k = 0; k < NIN; k += 4) {
        int kk = k + half * 2;                    // lane's K pair within window
        v2f a  = *(const v2f*)(x     + arow * NIN + kk);
        v2f nz = *(const v2f*)(noise + arow * NIN + kk);
        a = a + nz;
        #pragma unroll
        for (int nt = 0; nt < NT_H; ++nt) {
            int ncol = nt * 16 + m;
            int ncl  = (ncol < HH) ? ncol : (HH - 1);   // pad cols never stored
            v2f b = *(const v2f*)(wih + (size_t)ncl * NIN + kk);
            c[nt] = __builtin_amdgcn_wmma_f32_16x16x4_f32(false, a, false, b,
                                                          (short)0, c[nt], false, false);
        }
    }
    #pragma unroll
    for (int nt = 0; nt < NT_H; ++nt) {
        int ncol = nt * 16 + m;
        if (ncol < HH) {
            #pragma unroll
            for (int v = 0; v < 8; ++v) {
                size_t r = (size_t)(r0 + v + half * 8);
                xp[r * HH + ncol] = c[nt][v];
            }
        }
    }
}

// ---------------------------------------------------------------------------
// Kernel 2: serial scan  h = tanh(xp_t + h @ W_hh^T)
// 8 blocks (16 batch rows) x 7 waves (one 16-col tile each); h in LDS as bf16
// (ping-pong, 1 barrier/step); bf16 WMMA 16x16x32 (K padded to 128, pads are
// exact zeros). xp for step t+1 is prefetched into registers while step t's
// WMMA/LDS/barrier executes, pulling global latency off the serial chain.
// In-place: reads xp from ht[], overwrites the same slot with h_t.
// ---------------------------------------------------------------------------
__global__ void __launch_bounds__(224)
rnn_scan(const unsigned short* __restrict__ whhb,
         float* __restrict__ ht) {
    __shared__ unsigned short hbuf[2][16][LDSROW];

    int lane = threadIdx.x & 31;
    int wave = threadIdx.x >> 5;                  // 0..6 = N tile
    int b0   = blockIdx.x * 16;                   // batch row tile
    int m    = lane & 15, half = lane >> 4;
    int ncol = wave * 16 + m;                     // 0..111
    bool live = (ncol < HH);

    // zero h (t=0 state) and padding columns in both buffers
    for (int i = threadIdx.x; i < 2 * 16 * LDSROW; i += 224)
        ((unsigned short*)hbuf)[i] = 0;

    // Preload loop-invariant B fragments: lane holds W_hh^T[kbase+half*8..+7] and [kbase+16+half*8..+7][ncol]
    BF16Frag bfr[4];
    #pragma unroll
    for (int kk = 0; kk < 4; ++kk) {
        const v4u* p = (const v4u*)(whhb + (size_t)ncol * KP + kk * 32 + half * 8);
        bfr[kk].u[0] = p[0];
        bfr[kk].u[1] = p[2];
    }

    // Preload xp(t=0)
    float xpv[8];
    #pragma unroll
    for (int v = 0; v < 8; ++v) {
        size_t g = ((size_t)(b0 + v + half * 8) * TT + 0) * HH + ncol;
        xpv[v] = live ? ht[g] : 0.0f;
    }
    __syncthreads();

    for (int t = 0; t < TT; ++t) {
        int p = t & 1, q = p ^ 1;

        // issue xp(t+1) loads early; they overlap the WMMA + LDS + barrier below
        float xnxt[8];
        if (t + 1 < TT) {
            #pragma unroll
            for (int v = 0; v < 8; ++v) {
                size_t g = ((size_t)(b0 + v + half * 8) * TT + (t + 1)) * HH + ncol;
                xnxt[v] = live ? ht[g] : 0.0f;
            }
        } else {
            #pragma unroll
            for (int v = 0; v < 8; ++v) xnxt[v] = 0.0f;
        }

        v8f c = {};
        #pragma unroll
        for (int kk = 0; kk < 4; ++kk) {          // A frag: two 8-half runs per lane
            BF16Frag a;
            a.u[0] = *(const v4u*)&hbuf[p][m][kk * 32 + half * 8];
            a.u[1] = *(const v4u*)&hbuf[p][m][kk * 32 + 16 + half * 8];
            c = __builtin_amdgcn_wmma_f32_16x16x32_bf16(false, a.v, false, bfr[kk].v,
                                                        (short)0, c, false, false);
        }

        #pragma unroll
        for (int v = 0; v < 8; ++v) {
            int mrow = v + half * 8;              // local row 0..15
            float hv = TANHF(c[v] + xpv[v]);      // pad cols: tanh(0)=0
            if (live) {
                size_t g = ((size_t)(b0 + mrow) * TT + t) * HH + ncol;
                ht[g] = hv;                       // overwrite xp slot with h_t
            }
            hbuf[q][mrow][ncol] = f2bf(hv);
            xpv[v] = xnxt[v];
        }
        __syncthreads();                          // single barrier (ping-pong)
    }
}

// ---------------------------------------------------------------------------
// Kernel 3: out[r][o] = tanh(sum_h h_t[r][h] * W_out[o][h])
// K=100 = 25 fp32 WMMAs; both n_out tiles fused per wave so h_t is read once.
// ---------------------------------------------------------------------------
__global__ void out_proj(const float* __restrict__ ht,
                         const float* __restrict__ wout,
                         float* __restrict__ out) {
    int lane = threadIdx.x & 31;
    int wave = threadIdx.x >> 5;
    int r0   = (blockIdx.x * 8 + wave) * 16;
    int m    = lane & 15, half = lane >> 4;
    size_t arow = (size_t)(r0 + m);

    v8f c[2] = {(v8f){}, (v8f){}};
    #pragma unroll
    for (int k = 0; k < HH; k += 4) {
        int kk = k + half * 2;
        v2f a = *(const v2f*)(ht + arow * HH + kk);
        #pragma unroll
        for (int nt = 0; nt < 2; ++nt) {
            v2f b = *(const v2f*)(wout + (size_t)(nt * 16 + m) * HH + kk);
            c[nt] = __builtin_amdgcn_wmma_f32_16x16x4_f32(false, a, false, b,
                                                          (short)0, c[nt], false, false);
        }
    }
    #pragma unroll
    for (int nt = 0; nt < 2; ++nt) {
        int ncol = nt * 16 + m;
        #pragma unroll
        for (int v = 0; v < 8; ++v) {
            size_t r = (size_t)(r0 + v + half * 8);
            out[r * NOUT + ncol] = TANHF(c[nt][v]);
        }
    }
}

// ---------------------------------------------------------------------------
extern "C" void kernel_launch(void* const* d_in, const int* in_sizes, int n_in,
                              void* d_out, int out_size, void* d_ws, size_t ws_size,
                              hipStream_t stream) {
    const float* x     = (const float*)d_in[0];   // [B,T,64]
    const float* noise = (const float*)d_in[1];   // [B,T,64]
    const float* wih   = (const float*)d_in[2];   // [100,64]
    const float* whh   = (const float*)d_in[3];   // [100,100]
    const float* wout  = (const float*)d_in[4];   // [32,100]

    float* out = (float*)d_out;                          // [B,T,32]
    float* ht  = out + (size_t)BB * TT * NOUT;           // [B,T,100] (xp staging, then h_t)
    unsigned short* whhb = (unsigned short*)d_ws;        // bf16 W_hh^T, [112][128]

    pack_whh_bf16<<<dim3((NT_H * 16 * KP + 255) / 256), 256, 0, stream>>>(whh, whhb);
    input_proj   <<<dim3((BB * TT) / (16 * 8)), 256, 0, stream>>>(x, noise, wih, ht);
    rnn_scan     <<<dim3(BB / 16), 224, 0, stream>>>(whhb, ht);
    out_proj     <<<dim3((BB * TT) / (16 * 8)), 256, 0, stream>>>(ht, wout, out);
}
